// SphConv_51677046505608
// MI455X (gfx1250) — matrix operand order repaired and checked by reading, not compiled, not run on hardware
//
#include <hip/hip_runtime.h>
#include <math.h>

typedef __attribute__((ext_vector_type(16))) _Float16 v16h;
typedef __attribute__((ext_vector_type(8)))  _Float16 v8h;
typedef __attribute__((ext_vector_type(8)))  float    v8f;

#define BATCH   16384
#define CIN     128
#define COUT    128
#define NK      45
#define NGRP    5
#define BT      16      // batches per workgroup (WMMA M)
#define IPAD    136     // 128 halves + 8 pad (272 B rows -> bank spread)
#define TPB     256     // 8 waves of 32

// y[b,o,k] = s_g/128 * sum_i x[b,i,k] * w[o,i,g],  g = group(k)
__global__ __launch_bounds__(TPB) void sphconv_wmma(
    const float* __restrict__ x,   // [BATCH][CIN][NK]
    const float* __restrict__ w,   // [COUT][CIN][NGRP]
    float* __restrict__ y)         // [BATCH][COUT][NK]
{
  extern __shared__ _Float16 smem[];
  _Float16* wl = smem;                       // [COUT][IPAD]  B-operand: wl[o][i]
  _Float16* xl = smem + COUT * IPAD;         // [<=17][BT][IPAD] A-operand: xl[kl][b][i]

  const int tid   = threadIdx.x;
  const int lane  = tid & 31;
  const int wave  = tid >> 5;
  const int b0    = blockIdx.x * BT;

  const int n     = lane & 15;               // A row M / B col N / D col N
  const int hi    = lane >> 4;               // lane half select
  const int obase = wave * 16;               // this wave's o-tile

  const int kofs[NGRP] = {0, 1, 6, 15, 28};
  const int ngs [NGRP] = {1, 5, 9, 13, 17};

  for (int g = 0; g < NGRP; ++g) {
    const int kof = kofs[g];
    const int ngk = ngs[g];

    // ---- stage W_g as f16: wl[o][i] = w[o,i,g] (L2-resident after first WG)
    for (int t = tid; t < COUT * CIN; t += TPB) {
      int o = t >> 7, i = t & 127;
      wl[o * IPAD + i] = (_Float16)w[(size_t)(o * CIN + i) * NGRP + g];
    }
    // ---- stage X block as f16: xl[kl][b][i]; global reads k-fastest (coalesced)
    const int tot = BT * CIN * ngk;
    for (int t = tid; t < tot; t += TPB) {
      int kl = t % ngk;
      int bi = t / ngk;
      int i  = bi & 127;
      int b  = bi >> 7;
      float v = x[(size_t)(b0 + b) * (CIN * NK) + i * NK + (kof + kl)];
      xl[(kl * BT + b) * IPAD + i] = (_Float16)v;
    }
    __syncthreads();

    const float sg = (float)(6.283185307179586 *
                     sqrt(12.566370614359172 / (4.0 * (double)g + 1.0)) / 128.0);

    // ---- hoist the group's entire B operand (K=128) into registers:
    //      4 x v16h = 32 VGPRs, invariant across all k of this group.
    // B (32x16 f16): lane col N=n, K run hi*16+0..15 (i-contiguous)
    const _Float16* brow = wl + (obase + n) * IPAD + hi * 16;
    v16h bmat[4];
    #pragma unroll
    for (int s = 0; s < 4; ++s) {
      v8h w0 = *(const v8h*)(brow + s * 32);
      v8h w1 = *(const v8h*)(brow + s * 32 + 8);
      bmat[s] = __builtin_shufflevector(w0, w1,
                  0,1,2,3,4,5,6,7,8,9,10,11,12,13,14,15);
    }

    for (int kl = 0; kl < ngk; ++kl) {
      // A (16x32 f16): lane m=n, K run hi*16+0..15 (i-contiguous)
      const _Float16* arow = xl + (kl * BT + n) * IPAD + hi * 16;

      v8f acc = {0.f, 0.f, 0.f, 0.f, 0.f, 0.f, 0.f, 0.f};
      #pragma unroll
      for (int s = 0; s < 4; ++s) {
        v8h a0 = *(const v8h*)(arow + s * 32);
        v8h a1 = *(const v8h*)(arow + s * 32 + 8);
        v16h a = __builtin_shufflevector(a0, a1,
                   0,1,2,3,4,5,6,7,8,9,10,11,12,13,14,15);
        acc = __builtin_amdgcn_wmma_f32_16x16x32_f16(
                  /*neg_a=*/false, a, /*neg_b=*/false, bmat[s],
                  /*c_mod=*/(short)0, acc, /*reuse_a=*/false, /*reuse_b=*/false);
      }

      // D (16x16 f32): VGPR r -> batch row r + hi*8, col N=n -> o
      float* yp = y + ((size_t)b0 * COUT + (obase + n)) * NK + (kof + kl);
      #pragma unroll
      for (int r = 0; r < 8; ++r) {
        int bm = r + hi * 8;
        yp[(size_t)bm * (COUT * NK)] = sg * acc[r];
      }
    }
    __syncthreads();
  }
}

extern "C" void kernel_launch(void* const* d_in, const int* in_sizes, int n_in,
                              void* d_out, int out_size, void* d_ws, size_t ws_size,
                              hipStream_t stream) {
  const float* x = (const float*)d_in[0];   // (16384, 128, 45) f32
  const float* w = (const float*)d_in[1];   // (128, 128, 5)    f32
  float* y = (float*)d_out;                 // (16384, 128, 45) f32

  size_t shmem = (size_t)(COUT * IPAD + 17 * BT * IPAD) * sizeof(_Float16); // 108,800 B
  dim3 grid(BATCH / BT);   // 1024 workgroups
  dim3 block(TPB);
  hipLaunchKernelGGL(sphconv_wmma, grid, block, shmem, stream, x, w, y);
}
